// ProteinStateReconstructor2D_41738492183090
// MI455X (gfx1250) — matrix operand order repaired and checked by reading, not compiled, not run on hardware
//
#include <hip/hip_runtime.h>

// ---------------- static problem config ----------------
#define NUM_GRAPHS       64
#define NODES_PER_GRAPH  2048
#define NTOT             (NUM_GRAPHS * NODES_PER_GRAPH)   // 131072
#define NODE_DIM         256
#define COND_DIM         64
#define POOL_DIM         128        // OUT_H*OUT_W = 16*8
#define MLP_H            256
#define BN_EPS           1e-5f

// GEMM tiling: grid 1024 blocks, each block = 128 rows x 256 cols,
// 8 waves, each wave = 16 rows x 64 cols, looped over 4 row-chunks of 32.
#define ROWS_PER_BLOCK   128
#define GEMM_BLOCKS      (NTOT / ROWS_PER_BLOCK)          // 1024

typedef __attribute__((ext_vector_type(16))) __bf16 v16bf;
typedef __attribute__((ext_vector_type(8)))  float  v8f;

// ---------------- helpers ----------------
__device__ __forceinline__ v8f wmma_bf16(v16bf a, v16bf b, v8f c) {
  return __builtin_amdgcn_wmma_f32_16x16x32_bf16(
      false, a, false, b, (short)0, c, false, false);
}

// Build hi/lo bf16 A-fragment from 2 groups of 8 consecutive f32.
__device__ __forceinline__ void cvt16(const float* __restrict__ p0,
                                      const float* __restrict__ p1,
                                      v16bf& hi, v16bf& lo) {
#pragma unroll
  for (int i = 0; i < 8; ++i) {
    float v  = p0[i];
    __bf16 h = (__bf16)v;
    hi[i] = h;
    lo[i] = (__bf16)(v - (float)h);
  }
#pragma unroll
  for (int i = 0; i < 8; ++i) {
    float v  = p1[i];
    __bf16 h = (__bf16)v;
    hi[8 + i] = h;
    lo[8 + i] = (__bf16)(v - (float)h);
  }
}

// ---------------- kernel 1: adaptive avg pool ----------------
// pooled[g][oh*8+ow] = mean over 128 rows x 32 cols of x.
__global__ __launch_bounds__(256)
void pool_kernel(const float* __restrict__ x, float* __restrict__ pooled) {
  __shared__ float s[256];
  int g  = blockIdx.x >> 4;
  int oh = blockIdx.x & 15;
  int c  = threadIdx.x;
  const float* p = x + ((size_t)(g * NODES_PER_GRAPH + oh * 128)) * NODE_DIM + c;
  float acc = 0.f;
#pragma unroll 8
  for (int r = 0; r < 128; ++r) acc += p[(size_t)r * NODE_DIM];
  s[c] = acc;
  __syncthreads();
  if (c < 8) {
    float t = 0.f;
#pragma unroll
    for (int i = 0; i < 32; ++i) t += s[c * 32 + i];
    pooled[g * POOL_DIM + oh * 8 + c] = t * (1.0f / 4096.0f);
  }
}

// ---------------- kernel 2: pack W1[0:64,:] into bf16 hi/lo WMMA B-fragments --
// Layout: idx = ((tile*2 + chunk)*32 + lane)*16 + e
//   value = W1[chunk*32 + (e>>3)*16 + (lane>>4)*8 + (e&7)][tile*16 + (lane&15)]
__global__ __launch_bounds__(256)
void packw_kernel(const float* __restrict__ W1,
                  __bf16* __restrict__ wh, __bf16* __restrict__ wl) {
  int idx  = blockIdx.x * 256 + threadIdx.x;     // 0..16383
  int e    = idx & 15;
  int lane = (idx >> 4) & 31;
  int c    = (idx >> 9) & 1;
  int t    = idx >> 10;
  int krow = c * 32 + (e >> 3) * 16 + (lane >> 4) * 8 + (e & 7);      // 0..63
  int col  = t * 16 + (lane & 15);               // 0..255
  float v  = W1[(size_t)krow * MLP_H + col];
  __bf16 h = (__bf16)v;
  wh[idx] = h;
  wl[idx] = (__bf16)(v - (float)h);
}

// ---------------- kernel 3: per-graph pooled term ----------------
// pterm[g][col] = b1[col] + sum_k pooled[g][k] * W1[64+k][col]
__global__ __launch_bounds__(256)
void pterm_kernel(const float* __restrict__ pooled,
                  const float* __restrict__ W1,
                  const float* __restrict__ b1,
                  float* __restrict__ pterm) {
  int g = blockIdx.x, col = threadIdx.x;
  float s = b1[col];
  const float* pg = pooled + g * POOL_DIM;
#pragma unroll 4
  for (int k = 0; k < POOL_DIM; ++k)
    s += pg[k] * W1[(size_t)(COND_DIM + k) * MLP_H + col];
  pterm[g * MLP_H + col] = s;
}

// ---------------- GEMM core (shared by pass1/pass2) ----------------
// Computes acc tile (16x16) for one output tile with bf16x3 emulation.
__device__ __forceinline__ v8f gemm_tile(const v16bf ah[2], const v16bf al[2],
                                         const __bf16* __restrict__ wh,
                                         const __bf16* __restrict__ wl,
                                         int tile, int lane) {
  v8f acc = {};
#pragma unroll
  for (int c = 0; c < 2; ++c) {
    const v16bf bh = *(const v16bf*)(wh + ((size_t)((tile * 2 + c) * 32 + lane) * 16));
    const v16bf bl = *(const v16bf*)(wl + ((size_t)((tile * 2 + c) * 32 + lane) * 16));
    acc = wmma_bf16(al[c], bh, acc);   // low-order terms first
    acc = wmma_bf16(ah[c], bl, acc);
    acc = wmma_bf16(ah[c], bh, acc);
  }
  return acc;
}

// ---------------- kernel 4: pass1 — GEMM + BN partial stats ----------------
__global__ __launch_bounds__(256)
void pass1_kernel(const float* __restrict__ cond,
                  const float* __restrict__ pterm,
                  const __bf16* __restrict__ wh,
                  const __bf16* __restrict__ wl,
                  float* __restrict__ psum, float* __restrict__ psq) {
  __shared__ float s_sum[256];
  __shared__ float s_sq[256];
  int tid = threadIdx.x;
  s_sum[tid] = 0.f; s_sq[tid] = 0.f;
  __syncthreads();

  int wave = tid >> 5, lane = tid & 31;
  int half = lane >> 4, m16 = lane & 15;
  int rowGroup = wave >> 2, colGroup = wave & 3;
  int g = blockIdx.x >> 4;                           // 16 blocks per graph
  const float* pt = pterm + g * MLP_H;

#pragma unroll 1
  for (int rc = 0; rc < 4; ++rc) {
    int row = blockIdx.x * ROWS_PER_BLOCK + rc * 32 + rowGroup * 16 + m16;
    v16bf ah[2], al[2];
#pragma unroll
    for (int c = 0; c < 2; ++c) {
      const float* p = cond + (size_t)row * COND_DIM + c * 32 + half * 8;
      cvt16(p, p + 16, ah[c], al[c]);
    }
#pragma unroll 1
    for (int t = 0; t < 4; ++t) {
      int tile = colGroup * 4 + t;
      v8f acc = gemm_tile(ah, al, wh, wl, tile, lane);
      int n = tile * 16 + m16;
      float padd = pt[n];
      float ls = 0.f, lq = 0.f;
#pragma unroll
      for (int r = 0; r < 8; ++r) {
        float h = acc[r] + padd;
        ls += h; lq += h * h;
      }
      atomicAdd(&s_sum[n], ls);
      atomicAdd(&s_sq[n], lq);
    }
  }
  __syncthreads();
  psum[(size_t)blockIdx.x * 256 + tid] = s_sum[tid];
  psq [(size_t)blockIdx.x * 256 + tid] = s_sq[tid];
}

// ---------------- kernel 5: stat reduction stage 1 (1024 -> 64 rows) -------
__global__ __launch_bounds__(256)
void reduce1_kernel(const float* __restrict__ psum,
                    const float* __restrict__ psq,
                    float* __restrict__ rsum, float* __restrict__ rsq) {
  int c = threadIdx.x;
  float s = 0.f, q = 0.f;
#pragma unroll
  for (int i = 0; i < 16; ++i) {
    size_t o = (size_t)(blockIdx.x * 16 + i) * 256 + c;
    s += psum[o];
    q += psq[o];
  }
  rsum[blockIdx.x * 256 + c] = s;
  rsq [blockIdx.x * 256 + c] = q;
}

// ---------------- kernel 6: finalize BN -> scale/shift ----------------
__global__ __launch_bounds__(256)
void finalize_kernel(const float* __restrict__ rsum,
                     const float* __restrict__ rsq,
                     const float* __restrict__ gamma,
                     const float* __restrict__ beta,
                     float* __restrict__ scale, float* __restrict__ shiftv) {
  int c = threadIdx.x;
  float s = 0.f, q = 0.f;
#pragma unroll
  for (int i = 0; i < 64; ++i) {
    s += rsum[i * 256 + c];
    q += rsq [i * 256 + c];
  }
  const float invN = 1.0f / (float)NTOT;
  float mean = s * invN;
  float var  = q * invN - mean * mean;
  float sc   = gamma[c] * rsqrtf(var + BN_EPS);
  scale[c]  = sc;
  shiftv[c] = beta[c] - mean * sc;
}

// ---------------- kernel 7: pass2 — GEMM, BN affine, GELU, x W2 ----------
__global__ __launch_bounds__(256)
void pass2_kernel(const float* __restrict__ cond,
                  const float* __restrict__ pterm,
                  const __bf16* __restrict__ wh,
                  const __bf16* __restrict__ wl,
                  const float* __restrict__ scale,
                  const float* __restrict__ shiftv,
                  const float* __restrict__ W2,
                  const float* __restrict__ b2,
                  float* __restrict__ out) {
  __shared__ float sg[32 * 257];                     // 32 rows x 256 cols, padded
  __shared__ float sW2[MLP_H * 3];
  __shared__ float sb2[3];
  int tid = threadIdx.x;
  // cache W2/b2 in LDS once per block
#pragma unroll
  for (int i = 0; i < 3; ++i) sW2[i * 256 + tid] = W2[i * 256 + tid];
  if (tid < 3) sb2[tid] = b2[tid];

  int wave = tid >> 5, lane = tid & 31;
  int half = lane >> 4, m16 = lane & 15;
  int rowGroup = wave >> 2, colGroup = wave & 3;
  int g = blockIdx.x >> 4;
  const float* pt = pterm + g * MLP_H;
  __syncthreads();

#pragma unroll 1
  for (int rc = 0; rc < 4; ++rc) {
    int row = blockIdx.x * ROWS_PER_BLOCK + rc * 32 + rowGroup * 16 + m16;
    v16bf ah[2], al[2];
#pragma unroll
    for (int c = 0; c < 2; ++c) {
      const float* p = cond + (size_t)row * COND_DIM + c * 32 + half * 8;
      cvt16(p, p + 16, ah[c], al[c]);
    }
#pragma unroll 1
    for (int t = 0; t < 4; ++t) {
      int tile = colGroup * 4 + t;
      v8f acc = gemm_tile(ah, al, wh, wl, tile, lane);
      int n = tile * 16 + m16;
      float padd = pt[n], sc = scale[n], sh = shiftv[n];
#pragma unroll
      for (int r = 0; r < 8; ++r) {
        float h  = acc[r] + padd;
        float hn = h * sc + sh;
        // exact GELU: 0.5*x*(1+erf(x/sqrt(2)))
        float gv = 0.5f * hn * (1.0f + erff(hn * 0.70710678118654752f));
        int lr = rowGroup * 16 + r + 8 * half;       // local row in chunk (0..31)
        sg[lr * 257 + n] = gv;
      }
    }
    __syncthreads();
    if (tid < 96) {
      int lr = tid / 3, j = tid % 3;
      float s = sb2[j];
#pragma unroll 4
      for (int col = 0; col < 256; ++col)
        s += sg[lr * 257 + col] * sW2[col * 3 + j];
      int grow = blockIdx.x * ROWS_PER_BLOCK + rc * 32 + lr;
      out[(size_t)grow * 3 + j] = s;
    }
    __syncthreads();
  }
}

// ---------------- launcher ----------------
extern "C" void kernel_launch(void* const* d_in, const int* in_sizes, int n_in,
                              void* d_out, int out_size, void* d_ws, size_t ws_size,
                              hipStream_t stream) {
  (void)in_sizes; (void)n_in; (void)out_size; (void)ws_size;
  const float* x     = (const float*)d_in[0];
  // d_in[1] = batch: contiguous equal segments, not needed at runtime
  const float* cond  = (const float*)d_in[2];
  const float* W1    = (const float*)d_in[3];
  const float* b1    = (const float*)d_in[4];
  const float* gamma = (const float*)d_in[5];
  const float* beta  = (const float*)d_in[6];
  const float* W2    = (const float*)d_in[7];
  const float* b2    = (const float*)d_in[8];
  float* out = (float*)d_out;

  // workspace layout (floats first, then bf16 fragment arrays; 32B aligned)
  float* wsf    = (float*)d_ws;
  float* pooled = wsf;                              // 64*128 =  8192
  float* pterm  = pooled + NUM_GRAPHS * POOL_DIM;   // 64*256 = 16384
  float* scale  = pterm + NUM_GRAPHS * MLP_H;       // 256
  float* shiftv = scale + 256;                      // 256
  float* rsum   = shiftv + 256;                     // 64*256 = 16384
  float* rsq    = rsum + 64 * 256;                  // 16384
  float* psum   = rsq + 64 * 256;                   // 1024*256 = 262144
  float* psq    = psum + GEMM_BLOCKS * 256;         // 262144
  __bf16* wh    = (__bf16*)(psq + GEMM_BLOCKS * 256); // 16384 bf16
  __bf16* wl    = wh + 16 * 2 * 32 * 16;              // 16384 bf16

  pool_kernel    <<<NUM_GRAPHS * 16, 256, 0, stream>>>(x, pooled);
  packw_kernel   <<<64, 256, 0, stream>>>(W1, wh, wl);
  pterm_kernel   <<<NUM_GRAPHS, 256, 0, stream>>>(pooled, W1, b1, pterm);
  pass1_kernel   <<<GEMM_BLOCKS, 256, 0, stream>>>(cond, pterm, wh, wl, psum, psq);
  reduce1_kernel <<<64, 256, 0, stream>>>(psum, psq, rsum, rsq);
  finalize_kernel<<<1, 256, 0, stream>>>(rsum, rsq, gamma, beta, scale, shiftv);
  pass2_kernel   <<<GEMM_BLOCKS, 256, 0, stream>>>(cond, pterm, wh, wl, scale, shiftv,
                                                   W2, b2, out);
}
